// Informer_22265110463316
// MI455X (gfx1250) — matrix operand order repaired and checked by reading, not compiled, not run on hardware
//
#include <hip/hip_runtime.h>
#include <math.h>

// ---------------------------------------------------------------------------
// Informer forward for MI455X (gfx1250, wave32).
// Dense [M,64]x[64,64] linears -> v_wmma_f32_16x16x32_f16 (f32 accumulate),
// fully unguarded (M is always a multiple of 16, N=K=64), vectorized b128
// loads for the A fragment. Attention inner loops (Dh=8) are register-
// resident VALU flash loops. Everything lives in L2 (192 MB >> ~20 MB ws).
// ---------------------------------------------------------------------------

typedef _Float16 half16 __attribute__((ext_vector_type(16)));
typedef float    float8 __attribute__((ext_vector_type(8)));

#define DEV_INLINE __device__ __forceinline__

constexpr int BB = 2;      // batch
constexpr int HH = 8;      // heads
constexpr int DH = 8;      // head dim
constexpr int DM = 64;     // d_model == d_ff
constexpr float NEG_INF = -3.402823466e38f;

static inline int ceil_div(int a, int b) { return (a + b - 1) / b; }

DEV_INLINE float gelu_exact(float x) {
  return 0.5f * x * (1.0f + erff(x * 0.70710678118654752f));
}

DEV_INLINE unsigned hash_rng(unsigned l, unsigned s, unsigned seed) {
  unsigned r = l * 2654435761u + s * 40503u + seed * 668265263u + 0x9e3779b9u;
  r ^= r >> 13; r *= 2246822519u; r ^= r >> 16;
  return r;
}

// ---------------------------------------------------------------------------
// WMMA GEMM specialized to N = K = 64:  Y[M,64] = act(X[M,64] @ W[64,64] + b) + resid
// Requires M % 16 == 0 (true for every call site: 8640 / 4320 / 2160).
// One wave per 16x16 output tile, K consumed 32 at a time, no bounds checks
// anywhere -> straight-line vmem + cvt + wmma. resid==Y aliasing is safe.
// act: 0 = none, 1 = exact gelu.
// ---------------------------------------------------------------------------
__global__ void k_gemm_wmma(const float* __restrict__ X, const float* __restrict__ W,
                            const float* __restrict__ bias, const float* __restrict__ resid,
                            float* __restrict__ Y, int M, int act)
{
  const int wave = blockIdx.x * (blockDim.x >> 5) + (threadIdx.x >> 5);
  if (wave >= (M >> 4) * 4) return;        // wave-uniform: EXEC all-ones at WMMA
  const int tm = (wave >> 2) << 4;
  const int tn = (wave &  3) << 4;
  const int lane = threadIdx.x & 31;
  const int hs = lane >> 4;                // half-wave select
  const int r  = lane & 15;

  // A: row tm+r, K-chunks [hs*8 .. hs*8+7] and [+16 .. +23]  (contiguous f32)
  const float* ap = X + (tm + r) * DM + hs * 8;
  // B: column tn+r, K rows [hs*8 .. hs*8+7] and [+16 .. +23] (stride 64 f32)
  const float* bp = W + (hs * 8) * DM + (tn + r);

  float8 c = {};
#pragma unroll
  for (int k0 = 0; k0 < 64; k0 += 32) {
    const float4 a0 = *(const float4*)(ap + k0);
    const float4 a1 = *(const float4*)(ap + k0 + 4);
    const float4 a2 = *(const float4*)(ap + k0 + 16);
    const float4 a3 = *(const float4*)(ap + k0 + 20);
    half16 a, b;
    a[0]  = (_Float16)a0.x; a[1]  = (_Float16)a0.y; a[2]  = (_Float16)a0.z; a[3]  = (_Float16)a0.w;
    a[4]  = (_Float16)a1.x; a[5]  = (_Float16)a1.y; a[6]  = (_Float16)a1.z; a[7]  = (_Float16)a1.w;
    a[8]  = (_Float16)a2.x; a[9]  = (_Float16)a2.y; a[10] = (_Float16)a2.z; a[11] = (_Float16)a2.w;
    a[12] = (_Float16)a3.x; a[13] = (_Float16)a3.y; a[14] = (_Float16)a3.z; a[15] = (_Float16)a3.w;
#pragma unroll
    for (int j = 0; j < 16; ++j)
      b[j] = (_Float16)bp[(k0 + (j & 7) + ((j >> 3) << 4)) * DM];
    c = __builtin_amdgcn_wmma_f32_16x16x32_f16(false, a, false, b, (short)0, c, false, false);
  }

  // D layout: lane<16 -> col=lane, rows tm+0..7 ; lane>=16 -> col=lane-16, rows tm+8..15
  const int col = tn + r;
  const int row0 = tm + hs * 8;
  const float bv = bias ? bias[col] : 0.0f;
  float* yp = Y + row0 * DM + col;
  const float* rp = resid ? (resid + row0 * DM + col) : nullptr;
#pragma unroll
  for (int v = 0; v < 8; ++v) {
    float y = c[v] + bv;
    if (act == 1) y = gelu_exact(y);
    if (rp) y += rp[v * DM];
    yp[v * DM] = y;
  }
}

// ---------------------------------------------------------------------------
// Embedding: circular conv3 (Cin -> 64, no bias) + sinusoidal positional.
// ---------------------------------------------------------------------------
__global__ void k_embed(const float* __restrict__ x, const float* __restrict__ Wc,
                        float* __restrict__ out, int Lx, int Cin)
{
  const int i = blockIdx.x * blockDim.x + threadIdx.x;
  const int total = BB * Lx * DM;
  if (i >= total) return;
  const int d = i & (DM - 1);
  const int l = (i >> 6) % Lx;
  const int b = i / (Lx * DM);

  float acc = 0.0f;
#pragma unroll
  for (int k = 0; k < 3; ++k) {
    int sl = l - 1 + k;
    sl = (sl + Lx) % Lx;                   // circular pad
    const float* xr = x + (b * Lx + sl) * Cin;
    for (int cidx = 0; cidx < Cin; ++cidx)
      acc += xr[cidx] * Wc[(k * Cin + cidx) * DM + d];
  }
  const int pair = d >> 1;
  const float freq = expf(-(float)(2 * pair) * (logf(10000.0f) / (float)DM));
  const float ang = (float)l * freq;
  acc += (d & 1) ? cosf(ang) : sinf(ang);
  out[i] = acc;
}

// ---------------------------------------------------------------------------
// LayerNorm over width 64, one wave per row, in-place. Shuffle reductions.
// ---------------------------------------------------------------------------
__global__ void k_layernorm(float* __restrict__ X, const float* __restrict__ g,
                            const float* __restrict__ bta, int M)
{
  const int row = blockIdx.x * (blockDim.x >> 5) + (threadIdx.x >> 5);
  const int lane = threadIdx.x & 31;
  if (row >= M) return;
  float v0 = X[row * DM + lane];
  float v1 = X[row * DM + 32 + lane];
  float s = v0 + v1;
  for (int off = 16; off; off >>= 1) s += __shfl_xor(s, off);
  const float mean = s * (1.0f / 64.0f);
  const float d0 = v0 - mean, d1 = v1 - mean;
  float q = d0 * d0 + d1 * d1;
  for (int off = 16; off; off >>= 1) q += __shfl_xor(q, off);
  const float inv = rsqrtf(q * (1.0f / 64.0f) + 1e-5f);
  X[row * DM + lane]      = d0 * inv * g[lane]      + bta[lane];
  X[row * DM + 32 + lane] = d1 * inv * g[lane + 32] + bta[lane + 32];
}

// ---------------------------------------------------------------------------
// Prob-sparse attention helpers. Q/K/V are flat [B*L, 64]; head h occupies
// columns h*8..h*8+7. ctx layout: [B, H, L, 8] contiguous.
// ---------------------------------------------------------------------------
__global__ void k_vmean(const float* __restrict__ V, float* __restrict__ vm, int Lc)
{
  const int t = threadIdx.x;               // 128 threads = (b,h,d)
  if (t >= BB * HH * DH) return;
  const int b = t >> 6, h = (t >> 3) & 7, d = t & 7;
  float s = 0.0f;
  for (int l = 0; l < Lc; ++l) s += V[(b * Lc + l) * DM + h * DH + d];
  vm[t] = s / (float)Lc;
}

__global__ void k_ctx_bcast(const float* __restrict__ vm, float* __restrict__ ctx, int Lc)
{
  const int i = blockIdx.x * blockDim.x + threadIdx.x;
  const int total = BB * HH * Lc * DH;
  if (i >= total) return;
  const int d = i & 7;
  const int bh = (i >> 3) / Lc;
  ctx[i] = vm[bh * DH + d];
}

__global__ void k_cumsum_ctx(const float* __restrict__ V, float* __restrict__ ctx, int Lc)
{
  const int t = threadIdx.x;
  if (t >= BB * HH * DH) return;
  const int b = t >> 6, h = (t >> 3) & 7, d = t & 7;
  float s = 0.0f;
  for (int l = 0; l < Lc; ++l) {
    s += V[(b * Lc + l) * DM + h * DH + d];
    ctx[((b * HH + h) * Lc + l) * DH + d] = s;
  }
}

// Sparsity statistic M_l = max_s <Q_l, K_samp> - sum_s <Q_l, K_samp> / L_K
__global__ void k_probM(const float* __restrict__ Qf, const float* __restrict__ Kf,
                        float* __restrict__ Mout, int Lc, int nsamp, unsigned seed)
{
  const int i = blockIdx.x * blockDim.x + threadIdx.x;
  const int total = BB * HH * Lc;
  if (i >= total) return;
  const int l = i % Lc;
  const int h = (i / Lc) % HH;
  const int b = i / (Lc * HH);
  float q[DH];
#pragma unroll
  for (int d = 0; d < DH; ++d) q[d] = Qf[(b * Lc + l) * DM + h * DH + d];
  float mx = NEG_INF, sm = 0.0f;
  for (int s = 0; s < nsamp; ++s) {
    const int ks = (int)(hash_rng((unsigned)l, (unsigned)s, seed) % (unsigned)Lc);
    const float* kr = Kf + (b * Lc + ks) * DM + h * DH;
    float dot = 0.0f;
#pragma unroll
    for (int d = 0; d < DH; ++d) dot += q[d] * kr[d];
    mx = fmaxf(mx, dot);
    sm += dot;
  }
  Mout[i] = mx - sm / (float)Lc;
}

// Iterative top-u argmax per (b,h); destroys Mv (scratch). top stride 64.
__global__ void k_topk(float* __restrict__ Mv, int* __restrict__ top, int Lc, int u)
{
  const int bh = blockIdx.x;
  float* m = Mv + bh * Lc;
  __shared__ float sv[256];
  __shared__ int   si[256];
  const int t = threadIdx.x;
  for (int it = 0; it < u; ++it) {
    float best = NEG_INF; int bi = 0;
    for (int l = t; l < Lc; l += 256) {
      const float v = m[l];
      if (v > best) { best = v; bi = l; }
    }
    sv[t] = best; si[t] = bi;
    __syncthreads();
    for (int off = 128; off; off >>= 1) {
      if (t < off && sv[t + off] > sv[t]) { sv[t] = sv[t + off]; si[t] = si[t + off]; }
      __syncthreads();
    }
    if (t == 0) { top[bh * 64 + it] = si[0]; m[si[0]] = NEG_INF; }
    __syncthreads();
  }
}

// For each selected query: softmax(Q.K^T / sqrt(8)) @ V, scattered into ctx.
// One wave per (b,h,ui); lanes stride over keys; two-pass online softmax.
__global__ void k_prob_update(const float* __restrict__ Qf, const float* __restrict__ Kf,
                              const float* __restrict__ Vf, const int* __restrict__ top,
                              float* __restrict__ ctx, int Lc, int u, int masked)
{
  const int wave = blockIdx.x * (blockDim.x >> 5) + (threadIdx.x >> 5);
  const int total = BB * HH * u;
  if (wave >= total) return;
  const int lane = threadIdx.x & 31;
  const int ui = wave % u;
  const int h  = (wave / u) % HH;
  const int b  = wave / (u * HH);
  const int qidx = top[(b * HH + h) * 64 + ui];
  const float scale = 0.35355339059327373f;   // 1/sqrt(8)

  float q[DH];
#pragma unroll
  for (int d = 0; d < DH; ++d) q[d] = Qf[(b * Lc + qidx) * DM + h * DH + d];

  // pass 1: per-lane online (max, sumexp), then merge across the wave
  float m = NEG_INF, z = 0.0f;
  for (int s = lane; s < Lc; s += 32) {
    if (masked && s > qidx) continue;
    const float* kr = Kf + (b * Lc + s) * DM + h * DH;
    float sc = 0.0f;
#pragma unroll
    for (int d = 0; d < DH; ++d) sc += q[d] * kr[d];
    sc *= scale;
    const float nm = fmaxf(m, sc);
    z = z * expf(m - nm) + expf(sc - nm);
    m = nm;
  }
  for (int off = 16; off; off >>= 1) {
    const float m2 = __shfl_xor(m, off);
    const float z2 = __shfl_xor(z, off);
    const float nm = fmaxf(m, m2);
    float zn = 0.0f;
    if (m  > -1e37f) zn += z  * expf(m  - nm);
    if (m2 > -1e37f) zn += z2 * expf(m2 - nm);
    m = nm; z = zn;
  }
  const float invZ = (z > 0.0f) ? (1.0f / z) : 0.0f;

  // pass 2: weighted V accumulation
  float acc[DH];
#pragma unroll
  for (int d = 0; d < DH; ++d) acc[d] = 0.0f;
  for (int s = lane; s < Lc; s += 32) {
    if (masked && s > qidx) continue;
    const float* kr = Kf + (b * Lc + s) * DM + h * DH;
    float sc = 0.0f;
#pragma unroll
    for (int d = 0; d < DH; ++d) sc += q[d] * kr[d];
    const float wgt = expf(sc * scale - m) * invZ;
    const float* vr = Vf + (b * Lc + s) * DM + h * DH;
#pragma unroll
    for (int d = 0; d < DH; ++d) acc[d] += wgt * vr[d];
  }
#pragma unroll
  for (int d = 0; d < DH; ++d)
    for (int off = 16; off; off >>= 1) acc[d] += __shfl_xor(acc[d], off);
  if (lane == 0) {
#pragma unroll
    for (int d = 0; d < DH; ++d)
      ctx[((b * HH + h) * Lc + qidx) * DH + d] = acc[d];
  }
}

// ctx [B,H,L,8] -> out [B,L,64]. mix=0: gather heads into columns.
// mix=1: transpose(0,2,1,3)+reshape == flat copy of ctx.
__global__ void k_ctx_out(const float* __restrict__ ctx, float* __restrict__ out,
                          int Lc, int mix)
{
  const int i = blockIdx.x * blockDim.x + threadIdx.x;
  const int total = BB * Lc * DM;
  if (i >= total) return;
  if (mix) { out[i] = ctx[i]; return; }
  const int b = i / (Lc * DM);
  const int rem = i % (Lc * DM);
  const int l = rem / DM;
  const int cidx = rem & (DM - 1);
  const int h = cidx >> 3, d = cidx & 7;
  out[i] = ctx[((b * HH + h) * Lc + l) * DH + d];
}

// ---------------------------------------------------------------------------
// Full cross-attention, flash style: one thread per (b,h,q), online softmax
// over S keys, Dh=8 accumulators in registers. Output already [B,Lq,64].
// ---------------------------------------------------------------------------
__global__ void k_flash_xattn(const float* __restrict__ Qf, const float* __restrict__ Kf,
                              const float* __restrict__ Vf, float* __restrict__ out,
                              int Lq, int S)
{
  const int i = blockIdx.x * blockDim.x + threadIdx.x;
  const int total = BB * HH * Lq;
  if (i >= total) return;
  const int q = i % Lq;
  const int h = (i / Lq) % HH;
  const int b = i / (Lq * HH);
  const float scale = 0.35355339059327373f;

  float qv[DH];
#pragma unroll
  for (int d = 0; d < DH; ++d) qv[d] = Qf[(b * Lq + q) * DM + h * DH + d];

  float m = NEG_INF, z = 0.0f, acc[DH];
#pragma unroll
  for (int d = 0; d < DH; ++d) acc[d] = 0.0f;

  for (int s = 0; s < S; ++s) {
    const float* kr = Kf + (b * S + s) * DM + h * DH;
    float sc = 0.0f;
#pragma unroll
    for (int d = 0; d < DH; ++d) sc += qv[d] * kr[d];
    sc *= scale;
    const float nm = fmaxf(m, sc);
    const float r0 = expf(m - nm);
    const float w  = expf(sc - nm);
    z = z * r0 + w;
    const float* vr = Vf + (b * S + s) * DM + h * DH;
#pragma unroll
    for (int d = 0; d < DH; ++d) acc[d] = acc[d] * r0 + w * vr[d];
    m = nm;
  }
  const float invZ = (z > 0.0f) ? (1.0f / z) : 0.0f;
#pragma unroll
  for (int d = 0; d < DH; ++d)
    out[(b * Lq + q) * DM + h * DH + d] = acc[d] * invZ;
}

// ---------------------------------------------------------------------------
// Distillation: circular conv3 (64->64) + batchnorm + ELU.
// ---------------------------------------------------------------------------
__global__ void k_distil_conv(const float* __restrict__ x, const float* __restrict__ Wc,
                              const float* __restrict__ bias,
                              const float* __restrict__ bnw, const float* __restrict__ bnb,
                              const float* __restrict__ bnm, const float* __restrict__ bnv,
                              float* __restrict__ y, int Lx)
{
  const int i = blockIdx.x * blockDim.x + threadIdx.x;
  const int total = BB * Lx * DM;
  if (i >= total) return;
  const int d = i & (DM - 1);
  const int l = (i >> 6) % Lx;
  const int b = i / (Lx * DM);
  float acc = bias[d];
#pragma unroll
  for (int k = 0; k < 3; ++k) {
    int sl = l - 1 + k;
    sl = (sl + Lx) % Lx;
    const float* xr = x + (b * Lx + sl) * DM;
    for (int cidx = 0; cidx < DM; ++cidx)
      acc += xr[cidx] * Wc[(k * DM + cidx) * DM + d];
  }
  acc = (acc - bnm[d]) * rsqrtf(bnv[d] + 1e-5f) * bnw[d] + bnb[d];
  y[i] = (acc > 0.0f) ? acc : (expf(acc) - 1.0f);   // ELU
}

// Max-pool window 3, stride 2, pad (1,1) along L.
__global__ void k_maxpool(const float* __restrict__ x, float* __restrict__ y,
                          int Lin, int Lout)
{
  const int i = blockIdx.x * blockDim.x + threadIdx.x;
  const int total = BB * Lout * DM;
  if (i >= total) return;
  const int d = i & (DM - 1);
  const int lo = (i >> 6) % Lout;
  const int b = i / (Lout * DM);
  float m = NEG_INF;
#pragma unroll
  for (int w = 0; w < 3; ++w) {
    const int sl = 2 * lo - 1 + w;
    if (sl >= 0 && sl < Lin) m = fmaxf(m, x[(b * Lin + sl) * DM + d]);
  }
  y[i] = m;
}

// Final projection (64 -> 2) + softmax over the 2 classes.
__global__ void k_proj_softmax(const float* __restrict__ X, const float* __restrict__ Wp,
                               const float* __restrict__ bp, float* __restrict__ out, int M)
{
  const int row = blockIdx.x * blockDim.x + threadIdx.x;
  if (row >= M) return;
  float o0 = bp[0], o1 = bp[1];
  const float* xr = X + row * DM;
  for (int c = 0; c < DM; ++c) {
    o0 += xr[c] * Wp[c * 2 + 0];
    o1 += xr[c] * Wp[c * 2 + 1];
  }
  const float mx = fmaxf(o0, o1);
  const float e0 = expf(o0 - mx), e1 = expf(o1 - mx);
  const float inv = 1.0f / (e0 + e1);
  out[row * 2 + 0] = e0 * inv;
  out[row * 2 + 1] = e1 * inv;
}

// ---------------------------------------------------------------------------
// Host orchestration
// ---------------------------------------------------------------------------
struct AttnP { const float *wq,*bq,*wk,*bk,*wv,*bv,*wo,*bo; };
struct EncL  { AttnP a; const float *ln1g,*ln1b,*ln2g,*ln2b,*w1,*b1,*w2,*b2; };
struct DistP { const float *W,*b,*bnw,*bnb,*bnm,*bnv; };
struct DecL  { AttnP s, x; const float *ln1g,*ln1b,*ln2g,*ln2b,*ln3g,*ln3b,*w1,*b1,*w2,*b2; };

extern "C" void kernel_launch(void* const* d_in, const int* in_sizes, int n_in,
                              void* d_out, int out_size, void* d_ws, size_t ws_size,
                              hipStream_t stream)
{
  (void)in_sizes; (void)n_in; (void)out_size; (void)ws_size;
  const int Lfull = 4320;
  const int ENCIN = 12;

  // --- walk flattened params (setup_inputs insertion order, depth first) ---
  int ii = 0;
  auto nf = [&]() { return (const float*)d_in[ii++]; };
  const float* x_enc   = nf();
  const float* x_dec   = nf();
  const float* enc_emb = nf();
  const float* dec_emb = nf();
  EncL enc[3];
  for (int i = 0; i < 3; ++i) {
    AttnP a; a.wq=nf(); a.bq=nf(); a.wk=nf(); a.bk=nf(); a.wv=nf(); a.bv=nf(); a.wo=nf(); a.bo=nf();
    enc[i].a = a;
    enc[i].ln1g=nf(); enc[i].ln1b=nf(); enc[i].ln2g=nf(); enc[i].ln2b=nf();
    enc[i].w1=nf(); enc[i].b1=nf(); enc[i].w2=nf(); enc[i].b2=nf();
  }
  DistP dist[2];
  for (int i = 0; i < 2; ++i) {
    dist[i].W=nf(); dist[i].b=nf(); dist[i].bnw=nf(); dist[i].bnb=nf(); dist[i].bnm=nf(); dist[i].bnv=nf();
  }
  DecL dec[2];
  for (int i = 0; i < 2; ++i) {
    AttnP s; s.wq=nf(); s.bq=nf(); s.wk=nf(); s.bk=nf(); s.wv=nf(); s.bv=nf(); s.wo=nf(); s.bo=nf();
    AttnP x; x.wq=nf(); x.bq=nf(); x.wk=nf(); x.bk=nf(); x.wv=nf(); x.bv=nf(); x.wo=nf(); x.bo=nf();
    dec[i].s = s; dec[i].x = x;
    dec[i].ln1g=nf(); dec[i].ln1b=nf(); dec[i].ln2g=nf(); dec[i].ln2b=nf();
    dec[i].ln3g=nf(); dec[i].ln3b=nf();
    dec[i].w1=nf(); dec[i].b1=nf(); dec[i].w2=nf(); dec[i].b2=nf();
  }
  const float* enc_ng = nf(); const float* enc_nb = nf();
  const float* dec_ng = nf(); const float* dec_nb = nf();
  const float* proj_w = nf(); const float* proj_b = nf();

  // --- workspace carve-up ---
  const size_t SB = (size_t)BB * Lfull * DM;     // 552960 floats
  float* w = (float*)d_ws;
  size_t off = 0;
  auto alloc = [&](size_t n) { float* p = w + off; off += n; return p; };
  float* bQ  = alloc(SB);
  float* bK  = alloc(SB);
  float* bV  = alloc(SB);
  float* bT  = alloc(SB);   // reshaped attention output / flash output
  float* bA  = alloc(SB);   // stream ping
  float* bB  = alloc(SB);   // stream pong (holds encoder output at the end)
  float* bC  = alloc(SB);   // decoder stream
  float* bF  = alloc(SB);   // ffn mid / distil conv scratch
  float* CTX = alloc(SB);   // [B,H,L,8]
  float* Mb  = alloc((size_t)BB * HH * Lfull);
  float* VM  = alloc(BB * HH * DH);
  int*   TOP = (int*)alloc(BB * HH * 64);

  // --- launch helpers ---
  auto gemm = [&](const float* X, const float* Wm, const float* bias, const float* resid,
                  float* Y, int M, int act) {
    const int tiles = (M >> 4) * 4;          // N = 64 -> 4 column tiles
    const int wpb = 8;                       // 8 waves / 256 threads per block
    k_gemm_wmma<<<ceil_div(tiles, wpb), wpb * 32, 0, stream>>>(X, Wm, bias, resid, Y, M, act);
  };
  auto lnorm = [&](float* Xb, const float* g, const float* bb, int M) {
    k_layernorm<<<ceil_div(M, 8), 256, 0, stream>>>(Xb, g, bb, M);
  };
  auto prob_attn = [&](const float* act, const AttnP& p, int Lc, int masked, int mix,
                       unsigned seed, float* outbuf) {
    const int M = BB * Lc;
    gemm(act, p.wq, p.bq, nullptr, bQ, M, 0);
    gemm(act, p.wk, p.bk, nullptr, bK, M, 0);
    gemm(act, p.wv, p.bv, nullptr, bV, M, 0);
    int u = (int)(5.0 * ceil(log((double)Lc)));
    if (u > Lc) u = Lc;
    if (!masked) {
      k_vmean<<<1, 128, 0, stream>>>(bV, VM, Lc);
      const int tot = BB * HH * Lc * DH;
      k_ctx_bcast<<<ceil_div(tot, 256), 256, 0, stream>>>(VM, CTX, Lc);
    } else {
      k_cumsum_ctx<<<1, 128, 0, stream>>>(bV, CTX, Lc);
    }
    k_probM<<<ceil_div(BB * HH * Lc, 256), 256, 0, stream>>>(bQ, bK, Mb, Lc, u, seed);
    k_topk<<<BB * HH, 256, 0, stream>>>(Mb, TOP, Lc, u);
    k_prob_update<<<ceil_div(BB * HH * u * 32, 256), 256, 0, stream>>>(bQ, bK, bV, TOP, CTX,
                                                                       Lc, u, masked);
    k_ctx_out<<<ceil_div(BB * Lc * DM, 256), 256, 0, stream>>>(CTX, outbuf, Lc, mix);
  };

  // ===================== Encoder =====================
  int Lc = Lfull;
  k_embed<<<ceil_div(BB * Lc * DM, 256), 256, 0, stream>>>(x_enc, enc_emb, bA, Lc, ENCIN);
  float* cur = bA;
  for (int i = 0; i < 3; ++i) {
    const int M = BB * Lc;
    float* other = (cur == bA) ? bB : bA;
    prob_attn(cur, enc[i].a, Lc, /*masked=*/0, /*mix=*/0, /*seed=*/(unsigned)(i + 1), bT);
    gemm(bT, enc[i].a.wo, enc[i].a.bo, cur, other, M, 0);
    lnorm(other, enc[i].ln1g, enc[i].ln1b, M);
    gemm(other, enc[i].w1, enc[i].b1, nullptr, bF, M, 1);     // gelu
    gemm(bF, enc[i].w2, enc[i].b2, other, other, M, 0);       // in-place resid: safe
    lnorm(other, enc[i].ln2g, enc[i].ln2b, M);
    cur = other;
    if (i < 2) {
      k_distil_conv<<<ceil_div(BB * Lc * DM, 256), 256, 0, stream>>>(
          cur, dist[i].W, dist[i].b, dist[i].bnw, dist[i].bnb, dist[i].bnm, dist[i].bnv, bF, Lc);
      const int Lout = (Lc - 1) / 2 + 1;
      float* pooled = (cur == bA) ? bB : bA;
      k_maxpool<<<ceil_div(BB * Lout * DM, 256), 256, 0, stream>>>(bF, pooled, Lc, Lout);
      cur = pooled;
      Lc = Lout;
    }
  }
  lnorm(cur, enc_ng, enc_nb, BB * Lc);
  const float* ENC = cur;          // traces to bB; decoder never writes bB
  const int Lenc = Lc;             // 1080

  // ===================== Decoder =====================
  const int Ld = Lfull;
  const int Md = BB * Ld;
  k_embed<<<ceil_div(Md * DM, 256), 256, 0, stream>>>(x_dec, dec_emb, bC, Ld, ENCIN);
  float* dcur = bC;
  for (int i = 0; i < 2; ++i) {
    float* other = (dcur == bC) ? bA : bC;
    // masked prob self-attention, mix=True
    prob_attn(dcur, dec[i].s, Ld, /*masked=*/1, /*mix=*/1, /*seed=*/(unsigned)(101 + i), bT);
    gemm(bT, dec[i].s.wo, dec[i].s.bo, dcur, other, Md, 0);
    lnorm(other, dec[i].ln1g, dec[i].ln1b, Md);
    // full cross-attention against encoder output
    gemm(other, dec[i].x.wq, dec[i].x.bq, nullptr, bQ, Md, 0);
    gemm(ENC,   dec[i].x.wk, dec[i].x.bk, nullptr, bK, BB * Lenc, 0);
    gemm(ENC,   dec[i].x.wv, dec[i].x.bv, nullptr, bV, BB * Lenc, 0);
    k_flash_xattn<<<ceil_div(BB * HH * Ld, 256), 256, 0, stream>>>(bQ, bK, bV, bT, Ld, Lenc);
    gemm(bT, dec[i].x.wo, dec[i].x.bo, other, other, Md, 0);  // in-place resid: safe
    lnorm(other, dec[i].ln2g, dec[i].ln2b, Md);
    // FFN
    gemm(other, dec[i].w1, dec[i].b1, nullptr, bF, Md, 1);    // gelu
    gemm(bF, dec[i].w2, dec[i].b2, other, other, Md, 0);
    lnorm(other, dec[i].ln3g, dec[i].ln3b, Md);
    dcur = other;
  }
  lnorm(dcur, dec_ng, dec_nb, Md);
  k_proj_softmax<<<ceil_div(Md, 256), 256, 0, stream>>>(dcur, proj_w, proj_b, (float*)d_out, Md);
}
